// LLM_27384711479457
// MI455X (gfx1250) — matrix operand order repaired and checked by reading, not compiled, not run on hardware
//
#include <hip/hip_runtime.h>
#include <hip/hip_bf16.h>
#include <math.h>

// ---------------- problem constants ----------------
constexpr int Bv  = 4;
constexpr int Tv  = 1024;
constexpr int Cd  = 768;          // model dim
constexpr int FF  = 3072;         // 4*C
constexpr int Ev  = 8;            // experts
constexpr int Kv  = 2;            // top-k
constexpr int Nv  = Bv * Tv;      // 4096 tokens
constexpr int CAP = 2048;         // per-expert capacity

typedef __attribute__((ext_vector_type(16))) _Float16 v16h;
typedef __attribute__((ext_vector_type(8)))  _Float16 v8h;
typedef __attribute__((ext_vector_type(8)))  float    v8f;

// ---------------- kernel 1: router ----------------
__global__ __launch_bounds__(256) void k_router(const float* __restrict__ x,
                                                const float* __restrict__ wg,
                                                int* __restrict__ topi,
                                                float* __restrict__ prob) {
    const int n    = blockIdx.x * 8 + (threadIdx.x >> 5);
    const int lane = threadIdx.x & 31;
    if (n >= Nv) return;
    const float* xr = x + (size_t)n * Cd;
    float acc[Ev];
#pragma unroll
    for (int e = 0; e < Ev; ++e) acc[e] = 0.f;
    for (int c = lane; c < Cd; c += 32) {
        float xv = xr[c];
#pragma unroll
        for (int e = 0; e < Ev; ++e) acc[e] += xv * wg[e * Cd + c];
    }
#pragma unroll
    for (int e = 0; e < Ev; ++e)
        for (int off = 16; off > 0; off >>= 1)
            acc[e] += __shfl_xor(acc[e], off, 32);
    if (lane == 0) {
        int e1 = 0; float v1 = acc[0];
#pragma unroll
        for (int e = 1; e < Ev; ++e) if (acc[e] > v1) { v1 = acc[e]; e1 = e; }
        int e2 = -1; float v2 = -INFINITY;
#pragma unroll
        for (int e = 0; e < Ev; ++e) if (e != e1 && acc[e] > v2) { v2 = acc[e]; e2 = e; }
        float ex  = __expf(v2 - v1);
        float den = 1.f + ex;
        topi[n]      = e1;
        topi[Nv + n] = e2;
        prob[n]      = 1.f / den;
        prob[Nv + n] = ex / den;
    }
}

// ---------------- kernel 2: deterministic capacity assignment ----------------
__global__ __launch_bounds__(256) void k_assign(const int* __restrict__ topi,
                                                const float* __restrict__ prob,
                                                int* __restrict__ slot,
                                                float* __restrict__ wcomb,
                                                int* __restrict__ tok) {
    const int lane = threadIdx.x & 31;
    const int e    = threadIdx.x >> 5;
    for (int s = lane; s < CAP; s += 32) tok[e * CAP + s] = -1;
    int count = 0;
    for (int a = 0; a < Kv * Nv; a += 32) {
        const int idx = a + lane;
        const int eid = topi[idx];
        const bool m  = (eid == e);
        unsigned bal  = (unsigned)__ballot(m);
        int prefix    = __popc(bal & ((1u << lane) - 1u));
        if (m) {
            int rank = count + prefix;
            if (rank < CAP) {
                slot[idx]           = rank;
                wcomb[idx]          = prob[idx];
                tok[e * CAP + rank] = idx & (Nv - 1);
            } else {
                slot[idx]  = -1;
                wcomb[idx] = 0.f;
            }
        }
        count += __popc(bal);
    }
}

// ---------------- kernel 3: weight transpose + f32->f16 ----------------
// src: [E][R][Ccols] f32 (k-major)  ->  dst: [E][Ccols][R] f16 (n-major)
__global__ __launch_bounds__(256) void k_tconv(const float* __restrict__ src,
                                               _Float16* __restrict__ dst,
                                               int R, int Ccols) {
    __shared__ float tile[32][33];
    const int e  = blockIdx.z;
    const int r0 = blockIdx.y * 32, c0 = blockIdx.x * 32;
    const float* s = src + (size_t)e * R * Ccols;
    _Float16*    d = dst + (size_t)e * R * Ccols;
    const int tx = threadIdx.x & 31, ty = threadIdx.x >> 5;   // 32 x 8
#pragma unroll
    for (int i = 0; i < 32; i += 8)
        tile[ty + i][tx] = s[(size_t)(r0 + ty + i) * Ccols + c0 + tx];
    __syncthreads();
#pragma unroll
    for (int i = 0; i < 32; i += 8)
        d[(size_t)(c0 + ty + i) * R + r0 + tx] = (_Float16)tile[tx][ty + i];
}

// ---------------- kernel 4: token gather (dispatch) f32->f16 ----------------
__global__ __launch_bounds__(256) void k_gather(const float* __restrict__ x,
                                                const int* __restrict__ tok,
                                                _Float16* __restrict__ Xg) {
    const int sid   = blockIdx.x;          // e*CAP + slot
    const int token = tok[sid];
    _Float16* d = Xg + (size_t)sid * Cd;
    if (token >= 0) {
        const float* s = x + (size_t)token * Cd;
        for (int c = threadIdx.x; c < Cd; c += blockDim.x) d[c] = (_Float16)s[c];
    } else {
        for (int c = threadIdx.x; c < Cd; c += blockDim.x) d[c] = (_Float16)0.f;
    }
}

// ---------------- WMMA GEMM (shared by FC1 and FC2) ----------------
constexpr int TM = 128, TN = 128, TK = 32;
// 8 waves: 2 (M) x 4 (N); wave tile 64x32 -> 4x2 grid of 16x16 WMMA tiles.

__device__ __forceinline__ v16h cat16(v8h lo, v8h hi) {
    return __builtin_shufflevector(lo, hi, 0,1,2,3,4,5,6,7,8,9,10,11,12,13,14,15);
}

// Async copy 32 bytes (16 f16) global -> LDS.
// INST_OFFSET applies to both LDS and global addresses (ISA 08_async_tensor §4.4).
__device__ __forceinline__ void async_copy32(const _Float16* g, const _Float16* l) {
    unsigned           lds = (unsigned)(uintptr_t)l;   // low 32 bits = LDS byte addr
    unsigned long long ga  = (unsigned long long)(uintptr_t)g;
    asm volatile("global_load_async_to_lds_b128 %0, %1, off\n\t"
                 "global_load_async_to_lds_b128 %0, %1, off offset:16"
                 :: "v"(lds), "v"(ga) : "memory");
}

// A: [E][CAP][Kdim] f16 row-major. B: [E][Ndim][Kdim] f16 (pre-transposed, n-major).
// Double-buffered LDS pipeline: async ops retire in order, so waiting
// ASYNCcnt<=4 after issuing the next tile's 4 copies guarantees the current
// tile's 4 copies have landed.
template<int Kdim, int Ndim, bool GELU>
__global__ __launch_bounds__(256) void k_gemm(const _Float16* __restrict__ Aall,
                                              const _Float16* __restrict__ Ball,
                                              const float* __restrict__ bias,
                                              _Float16* __restrict__ outH,
                                              float* __restrict__ outF) {
    __shared__ __align__(16) _Float16 As[2][TM][40];   // [buf][row][k], padded
    __shared__ __align__(16) _Float16 Bs[2][TN][40];   // [buf][n][k],   padded
    const int e  = blockIdx.z;
    const int m0 = blockIdx.y * TM;
    const int n0 = blockIdx.x * TN;
    const int t    = threadIdx.x;
    const int lane = t & 31, wave = t >> 5;
    const int wm = wave & 1, wn = wave >> 1;

    const _Float16* A0 = Aall + (size_t)e * CAP  * Kdim + (size_t)m0 * Kdim;
    const _Float16* B0 = Ball + (size_t)e * Ndim * Kdim + (size_t)n0 * Kdim;

    // staging: thread -> (row, 16-k half); same pattern for A and B tiles
    const int srow = t >> 1, shalf = t & 1;
    const _Float16* ga = A0 + (size_t)srow * Kdim + shalf * 16;
    const _Float16* gb = B0 + (size_t)srow * Kdim + shalf * 16;

    v8f acc[4][2];
#pragma unroll
    for (int i = 0; i < 4; ++i)
#pragma unroll
        for (int j = 0; j < 2; ++j) acc[i][j] = (v8f){};

    constexpr int NT = Kdim / TK;
    // prologue: stage tile 0 into buffer 0
    async_copy32(ga, &As[0][srow][shalf * 16]);
    async_copy32(gb, &Bs[0][srow][shalf * 16]);

    for (int kt = 0; kt < NT; ++kt) {
        const int cur = kt & 1;
        if (kt + 1 < NT) {
            const int k1 = (kt + 1) * TK;
            async_copy32(ga + k1, &As[cur ^ 1][srow][shalf * 16]);
            async_copy32(gb + k1, &Bs[cur ^ 1][srow][shalf * 16]);
            asm volatile("s_wait_asynccnt 0x4" ::: "memory");  // current tile landed
        } else {
            asm volatile("s_wait_asynccnt 0x0" ::: "memory");  // last tile landed
        }
        __syncthreads();

        // ---- fragments per ISA 16-bit layouts ----
        const int kA = (lane >> 4) * 8;    // A: lanes 0-15 K0-7/16-23; 16-31 K8-15/24-31
        const int kB = (lane >> 4) * 8;    // B: lanes 0-15 K0-7/16-23; 16-31 K8-15/24-31
        v16h afr[4], bfr[2];
#pragma unroll
        for (int i = 0; i < 4; ++i) {
            const int r = wm * 64 + i * 16 + (lane & 15);
            afr[i] = cat16(*(const v8h*)&As[cur][r][kA], *(const v8h*)&As[cur][r][kA + 16]);
        }
#pragma unroll
        for (int j = 0; j < 2; ++j) {
            const int nc = wn * 32 + j * 16 + (lane & 15);
            bfr[j] = cat16(*(const v8h*)&Bs[cur][nc][kB], *(const v8h*)&Bs[cur][nc][kB + 16]);
        }
#pragma unroll
        for (int i = 0; i < 4; ++i)
#pragma unroll
            for (int j = 0; j < 2; ++j)
                acc[i][j] = __builtin_amdgcn_wmma_f32_16x16x32_f16(
                    false, afr[i], false, bfr[j], (short)0, acc[i][j], false, false);
        __syncthreads();   // protect buffer overwritten next iteration
    }

    // ---- epilogue ----
#pragma unroll
    for (int i = 0; i < 4; ++i)
#pragma unroll
        for (int j = 0; j < 2; ++j) {
            const int nc = n0 + wn * 32 + j * 16 + (lane & 15);
            const float b = bias[(size_t)e * Ndim + nc];
#pragma unroll
            for (int r = 0; r < 8; ++r) {
                const int m = m0 + wm * 64 + i * 16 + r + 8 * (lane >> 4);
                float v = acc[i][j][r] + b;
                if (GELU) {
                    v = 0.5f * v * (1.f + erff(v * 0.70710678118654752f));
                    outH[(size_t)(e * CAP + m) * Ndim + nc] = (_Float16)v;
                } else {
                    outF[(size_t)(e * CAP + m) * Ndim + nc] = v;
                }
            }
        }
}

// ---------------- kernel 5: combine ----------------
__global__ __launch_bounds__(256) void k_combine(const int* __restrict__ topi,
                                                 const int* __restrict__ slot,
                                                 const float* __restrict__ wcomb,
                                                 const float* __restrict__ expout,
                                                 float* __restrict__ out) {
    const int n  = blockIdx.x;
    const int s0 = slot[n],      e0 = topi[n];
    const int s1 = slot[Nv + n], e1 = topi[Nv + n];
    const float w0 = wcomb[n], w1 = wcomb[Nv + n];
    const float* r0 = (s0 >= 0) ? expout + (size_t)(e0 * CAP + s0) * Cd : nullptr;
    const float* r1 = (s1 >= 0) ? expout + (size_t)(e1 * CAP + s1) * Cd : nullptr;
    for (int c = threadIdx.x; c < Cd; c += blockDim.x) {
        float acc = 0.f;
        if (r0) acc += w0 * r0[c];
        if (r1) acc += w1 * r1[c];
        out[(size_t)n * Cd + c] = acc;
    }
}

// ---------------- launch ----------------
extern "C" void kernel_launch(void* const* d_in, const int* in_sizes, int n_in,
                              void* d_out, int out_size, void* d_ws, size_t ws_size,
                              hipStream_t stream) {
    const float* x     = (const float*)d_in[0];   // [B,T,C]
    const float* w_g   = (const float*)d_in[1];   // [E,C]
    const float* c_fc  = (const float*)d_in[2];   // [E,C,4C]
    const float* fcb   = (const float*)d_in[3];   // [E,1,4C]
    const float* c_prj = (const float*)d_in[4];   // [E,4C,C]
    const float* pb    = (const float*)d_in[5];   // [E,1,C]
    float* out = (float*)d_out;

    // workspace layout
    char* ws = (char*)d_ws;
    size_t off = 0;
    int*   topi  = (int*)(ws + off);   off += (size_t)Kv * Nv * sizeof(int);
    float* prob  = (float*)(ws + off); off += (size_t)Kv * Nv * sizeof(float);
    int*   slot  = (int*)(ws + off);   off += (size_t)Kv * Nv * sizeof(int);
    float* wcomb = (float*)(ws + off); off += (size_t)Kv * Nv * sizeof(float);
    int*   tok   = (int*)(ws + off);   off += (size_t)Ev * CAP * sizeof(int);
    off = (off + 255) & ~(size_t)255;
    _Float16* Xg  = (_Float16*)(ws + off); off += (size_t)Ev * CAP * Cd * sizeof(_Float16); // 25.2 MB
    _Float16* w1t = (_Float16*)(ws + off); off += (size_t)Ev * FF * Cd * sizeof(_Float16);  // 37.7 MB
    _Float16* w2t = (_Float16*)(ws + off); off += (size_t)Ev * Cd * FF * sizeof(_Float16);  // 37.7 MB
    _Float16* H   = (_Float16*)(ws + off); off += (size_t)Ev * CAP * FF * sizeof(_Float16); // 100.7 MB
    float* expout = (float*)(ws + off);    off += (size_t)Ev * CAP * Cd * sizeof(float);    // 50.3 MB
    (void)ws_size; (void)in_sizes; (void)n_in; (void)out_size;

    k_router <<<Nv / 8, 256, 0, stream>>>(x, w_g, topi, prob);
    k_assign <<<1, 256, 0, stream>>>(topi, prob, slot, wcomb, tok);
    // c_fc  [E][Cd][FF] -> w1t [E][FF][Cd]
    k_tconv  <<<dim3(FF / 32, Cd / 32, Ev), 256, 0, stream>>>(c_fc, w1t, Cd, FF);
    // c_proj [E][FF][Cd] -> w2t [E][Cd][FF]
    k_tconv  <<<dim3(Cd / 32, FF / 32, Ev), 256, 0, stream>>>(c_prj, w2t, FF, Cd);
    k_gather <<<Ev * CAP, 256, 0, stream>>>(x, tok, Xg);
    k_gemm<Cd, FF, true>  <<<dim3(FF / TN, CAP / TM, Ev), 256, 0, stream>>>(Xg, w1t, fcb, H, nullptr);
    k_gemm<FF, Cd, false> <<<dim3(Cd / TN, CAP / TM, Ev), 256, 0, stream>>>(H, w2t, pb, nullptr, expout);
    k_combine<<<Nv, 256, 0, stream>>>(topi, slot, wcomb, expout, out);
}
